// GraphAgent_84095459656235
// MI455X (gfx1250) — matrix operand order repaired and checked by reading, not compiled, not run on hardware
//
#include <hip/hip_runtime.h>
#include <hip/hip_bf16.h>
#include <math.h>

// ---------------------------------------------------------------------------
// MI455X (gfx1250) fused GNN edge-scoring pipeline.
// WMMA f32<-f16 16x16x32 everywhere a GEMM appears. Wave32.
// ---------------------------------------------------------------------------

typedef __attribute__((ext_vector_type(16))) _Float16 v16h;
typedef __attribute__((ext_vector_type(4)))  _Float16 h4;
typedef __attribute__((ext_vector_type(2)))  _Float16 h2;
typedef __attribute__((ext_vector_type(8)))  float    v8f;

#define NND   25000
#define NEDGE 400000
#define HH    128

static __device__ __forceinline__ v8f wmma_f16(v16h a, v16h b, v8f c) {
  // v_wmma_f32_16x16x32_f16  (neg_a, A, neg_b, B, c_mod, C, reuse_a, reuse_b)
  return __builtin_amdgcn_wmma_f32_16x16x32_f16(false, a, false, b, (short)0, c,
                                                false, false);
}

// A-matrix (16x32 f16) per-lane K offset: lanes 0-15 hold K pairs
// {0,2,4,6,16,18,20,22}; lanes 16-31 hold {8,10,12,14,24,26,28,30}.
static __device__ __forceinline__ int akoff(int i, int hh) {
  return ((i < 4) ? 2 * i : 2 * i + 8) + 8 * hh;
}

// ------------------------------- utility kernels ---------------------------

__global__ void k_zero(float* p, int n) {
  int i = blockIdx.x * 256 + threadIdx.x;
  if (i < n) p[i] = 0.0f;
}

__global__ void k_deg(const int* __restrict__ dst, float* __restrict__ deg) {
  int e = blockIdx.x * 256 + threadIdx.x;
  if (e < NEDGE) atomicAdd(&deg[dst[e]], 1.0f);
}

__global__ void k_dinv(const float* __restrict__ deg, float* __restrict__ dinv) {
  int i = blockIdx.x * 256 + threadIdx.x;
  if (i < NND) dinv[i] = rsqrtf(deg[i] + 1.0f);
}

// f32 -> f16 convert, 4 elements / thread (b128 load, b64 store)
__global__ void k_f32tof16(const float* __restrict__ in, _Float16* __restrict__ out,
                           int n4) {
  int i = blockIdx.x * 256 + threadIdx.x;
  if (i < n4) {
    float4 v = ((const float4*)in)[i];
    h4 o;
    o[0] = (_Float16)v.x; o[1] = (_Float16)v.y;
    o[2] = (_Float16)v.z; o[3] = (_Float16)v.w;
    ((h4*)out)[i] = o;
  }
}

// Fold small linear pieces:
//   wef[16x128]  = We @ Wr[2H:]           (edge_attr path folded through Wr2)
//   biasp[128]   = be @ Wr[2H:] + br
//   cvec[128]    = graph_rep @ Wp1[:H] + subgraph_rep @ Wp1[H:2H] + bp1
__global__ void k_setup(const float* __restrict__ We, const float* __restrict__ Wr,
                        const float* __restrict__ br, const float* __restrict__ be,
                        const float* __restrict__ grep, const float* __restrict__ srep,
                        const float* __restrict__ Wp1, const float* __restrict__ bp1,
                        float* __restrict__ wef, float* __restrict__ biasp,
                        float* __restrict__ cvec) {
  int t = blockIdx.x * 256 + threadIdx.x;
  if (t < 16 * HH) {
    int r = t >> 7, n = t & 127;
    float s = 0.0f;
    for (int k = 0; k < HH; ++k) s += We[r * HH + k] * Wr[(2 * HH + k) * HH + n];
    wef[t] = s;
  } else if (t < 16 * HH + HH) {
    int n = t - 16 * HH;
    float s = br[n];
    for (int k = 0; k < HH; ++k) s += be[k] * Wr[(2 * HH + k) * HH + n];
    biasp[n] = s;
  } else if (t < 16 * HH + 2 * HH) {
    int n = t - 16 * HH - HH;
    float s = bp1[n];
    for (int k = 0; k < HH; ++k)
      s += grep[k] * Wp1[k * HH + n] + srep[k] * Wp1[(HH + k) * HH + n];
    cvec[n] = s;
  }
}

// Pack a (Krows x 128) f32 row-major weight into WMMA B fragments (f16).
// Fragment (kt,nt): lane%16 = N, elements (2i,2i+1) = K pair akoff(i, lane/16).
// grid = Ktiles*8 blocks of 32 threads; blockIdx.x == kt*8+nt == fragment index.
__global__ void k_pack(const float* __restrict__ W, _Float16* __restrict__ out,
                       int Krows) {
  int kt = blockIdx.x >> 3, nt = blockIdx.x & 7, lane = threadIdx.x;
  int n = nt * 16 + (lane & 15);
  _Float16* o = out + ((size_t)blockIdx.x * 32 + lane) * 16;
#pragma unroll
  for (int i = 0; i < 8; ++i) {
    int K = kt * 32 + akoff(i, lane >> 4);
    float a = (K < Krows) ? W[K * HH + n] : 0.0f;
    float b = (K + 1 < Krows) ? W[(K + 1) * HH + n] : 0.0f;
    o[2 * i]     = (_Float16)a;
    o[2 * i + 1] = (_Float16)b;
  }
}

// ------------------------- GCN layer: node GEMM ----------------------------
// h = out_f16 @ W  (N x 128 @ 128 x 128), one wave per 16-row tile.
__global__ void __launch_bounds__(256) k_gemm_node(
    const _Float16* __restrict__ A, const _Float16* __restrict__ BW,
    float* __restrict__ Hout) {
  const int NT = (NND + 15) / 16;
  int wave = blockIdx.x * 8 + (threadIdx.x >> 5);
  if (wave >= NT) return;  // uniform per wave -> EXEC stays all-ones for WMMA
  int lane = threadIdx.x & 31, m = lane & 15, hh = lane >> 4;
  int row = wave * 16 + m;
  if (row >= NND) row = NND - 1;  // clamp gather; stores are predicated
  const _Float16* rp = A + (size_t)row * HH;
  const bool full = (wave * 16 + 16 <= NND);  // wave-uniform fast path

  v16h a[4];
#pragma unroll
  for (int kt = 0; kt < 4; ++kt)
#pragma unroll
    for (int i = 0; i < 8; ++i) {
      h2 t = *(const h2*)(rp + kt * 32 + akoff(i, hh));
      a[kt][2 * i] = t[0];
      a[kt][2 * i + 1] = t[1];
    }

#pragma unroll
  for (int nt = 0; nt < 8; ++nt) {
    v8f acc;
#pragma unroll
    for (int r = 0; r < 8; ++r) acc[r] = 0.0f;
#pragma unroll
    for (int kt = 0; kt < 4; ++kt) {
      v16h b = *(const v16h*)(BW + ((size_t)(kt * 8 + nt) * 32 + lane) * 16);
      acc = wmma_f16(a[kt], b, acc);
    }
    int col = nt * 16 + m;
    float* ob = Hout + (size_t)(wave * 16 + 8 * hh) * HH + col;
    if (full) {
#pragma unroll
      for (int r = 0; r < 8; ++r) ob[(size_t)r * HH] = acc[r];
    } else {
#pragma unroll
      for (int r = 0; r < 8; ++r) {
        int rr = wave * 16 + r + 8 * hh;
        if (rr < NND) Hout[(size_t)rr * HH + col] = acc[r];
      }
    }
  }
}

// agg[dst] += h[src] * dinv[src]*dinv[dst]   (one thread per (edge, 4 features))
__global__ void k_scatter(const float* __restrict__ Hm, const int* __restrict__ src,
                          const int* __restrict__ dst, const float* __restrict__ dinv,
                          float* __restrict__ agg) {
  int idx = blockIdx.x * 256 + threadIdx.x;
  if (idx < NEDGE * (HH / 4)) {
    int e = idx >> 5, j = (idx & 31) * 4;
    int s = src[e], d = dst[e];
    float w = dinv[s] * dinv[d];
    float4 hv = *(const float4*)(Hm + (size_t)s * HH + j);
    float* ap = agg + (size_t)d * HH + j;
    atomicAdd(ap + 0, hv.x * w);
    atomicAdd(ap + 1, hv.y * w);
    atomicAdd(ap + 2, hv.z * w);
    atomicAdd(ap + 3, hv.w * w);
  }
}

// out = relu(agg + h*dinv^2 + b)  -> f16 for the next GEMM / edge stage
// 4 elements / thread
__global__ void k_finish(const float* __restrict__ agg, const float* __restrict__ Hm,
                         const float* __restrict__ dinv, const float* __restrict__ bias,
                         _Float16* __restrict__ outf16) {
  int idx = blockIdx.x * 256 + threadIdx.x;
  if (idx < NND * (HH / 4)) {
    int i = idx >> 5, j = (idx & 31) * 4;
    float di = dinv[i];
    float sn = di * di;
    float4 av = ((const float4*)agg)[idx];
    float4 hv = ((const float4*)Hm)[idx];
    float4 bv = *(const float4*)(bias + j);
    h4 o;
    o[0] = (_Float16)fmaxf(av.x + hv.x * sn + bv.x, 0.0f);
    o[1] = (_Float16)fmaxf(av.y + hv.y * sn + bv.y, 0.0f);
    o[2] = (_Float16)fmaxf(av.z + hv.z * sn + bv.z, 0.0f);
    o[3] = (_Float16)fmaxf(av.w + hv.w * sn + bv.w, 0.0f);
    ((h4*)outf16)[idx] = o;
  }
}

// ---------------------- fused per-edge MLP (main kernel) -------------------
// One wave = 16 edges. er (16x128) stays in regs/LDS; never hits HBM.
__global__ void __launch_bounds__(256, 1) k_edge(
    const _Float16* __restrict__ outF, const float* __restrict__ eattr,
    const _Float16* __restrict__ Br0, const _Float16* __restrict__ Br1,
    const _Float16* __restrict__ Be, const _Float16* __restrict__ Bp,
    const float* __restrict__ biasp, const float* __restrict__ cvec,
    const float* __restrict__ wp2, const float* __restrict__ bp2,
    const int* __restrict__ src, const int* __restrict__ dst,
    const unsigned char* __restrict__ state, float* __restrict__ outp) {
  __shared__ _Float16 lds[8 * 16 * HH];  // 32 KB, wave-private slices
  int waveInB = threadIdx.x >> 5;
  int tile = blockIdx.x * 8 + waveInB;  // grid is exact: 3125*8 = 25000 tiles
  int lane = threadIdx.x & 31, m = lane & 15, hh = lane >> 4;
  int base = tile * 16;

  // Warm WGP$/L2 with the three 32KB stage weights (global_prefetch_b8).
  // 256 threads x 128B strides cover each matrix exactly once per block.
  {
    size_t po = (size_t)threadIdx.x * 128;
    __builtin_prefetch((const char*)Br0 + po, 0, 3);
    __builtin_prefetch((const char*)Br1 + po, 0, 3);
    __builtin_prefetch((const char*)Bp + po, 0, 3);
  }

  int sA = src[base + m];
  int dA = dst[base + m];
  const _Float16* sp = outF + (size_t)sA * HH;
  const _Float16* dp = outF + (size_t)dA * HH;

  // A fragments for gathered src/dst node rows (4 K-tiles each)
  v16h as[4], ad[4];
#pragma unroll
  for (int kt = 0; kt < 4; ++kt)
#pragma unroll
    for (int i = 0; i < 8; ++i) {
      int k = kt * 32 + akoff(i, hh);
      h2 ts = *(const h2*)(sp + k);
      h2 td = *(const h2*)(dp + k);
      as[kt][2 * i] = ts[0]; as[kt][2 * i + 1] = ts[1];
      ad[kt][2 * i] = td[0]; ad[kt][2 * i + 1] = td[1];
    }

  // edge_attr row (16 wide), zero-padded to K=32, converted to f16
  v16h ae;
  const float* ep = eattr + (size_t)(base + m) * 16;
#pragma unroll
  for (int i = 0; i < 8; ++i) {
    int k = akoff(i, hh);
    if (k < 16) {
      ae[2 * i]     = (_Float16)ep[k];
      ae[2 * i + 1] = (_Float16)ep[k + 1];
    } else {
      ae[2 * i] = (_Float16)0.0f;
      ae[2 * i + 1] = (_Float16)0.0f;
    }
  }

  _Float16* myLds = lds + waveInB * 16 * HH;

  // Stage 1: er = relu(src@Wr0 + dst@Wr1 + eattr@We' + b')  -> LDS (f16)
#pragma unroll
  for (int nt = 0; nt < 8; ++nt) {
    float bv = biasp[nt * 16 + m];
    v8f acc;
#pragma unroll
    for (int r = 0; r < 8; ++r) acc[r] = bv;
    {
      v16h b = *(const v16h*)(Be + ((size_t)nt * 32 + lane) * 16);
      acc = wmma_f16(ae, b, acc);
    }
#pragma unroll
    for (int kt = 0; kt < 4; ++kt) {
      v16h b0 = *(const v16h*)(Br0 + ((size_t)(kt * 8 + nt) * 32 + lane) * 16);
      acc = wmma_f16(as[kt], b0, acc);
      v16h b1 = *(const v16h*)(Br1 + ((size_t)(kt * 8 + nt) * 32 + lane) * 16);
      acc = wmma_f16(ad[kt], b1, acc);
    }
    // C layout: (vgpr r, lane) = element (M = r + 8*hh, N = nt*16 + m)
#pragma unroll
    for (int r = 0; r < 8; ++r) {
      float v = fmaxf(acc[r], 0.0f);
      myLds[(r + 8 * hh) * HH + nt * 16 + m] = (_Float16)v;
    }
  }

  // Reload er as A fragments (wave-private LDS; DS deps handled by s_wait_dscnt)
  v16h a2[4];
  const _Float16* lrow = myLds + m * HH;
#pragma unroll
  for (int kt = 0; kt < 4; ++kt)
#pragma unroll
    for (int i = 0; i < 8; ++i) {
      h2 t = *(const h2*)(lrow + kt * 32 + akoff(i, hh));
      a2[kt][2 * i] = t[0];
      a2[kt][2 * i + 1] = t[1];
    }

  // Stage 2+3: h1 = relu(cvec + er@Wp); partial dot with wp2 on the fly
  float partial[8];
#pragma unroll
  for (int r = 0; r < 8; ++r) partial[r] = 0.0f;
#pragma unroll
  for (int nt = 0; nt < 8; ++nt) {
    float cv = cvec[nt * 16 + m];
    v8f acc;
#pragma unroll
    for (int r = 0; r < 8; ++r) acc[r] = cv;
#pragma unroll
    for (int kt = 0; kt < 4; ++kt) {
      v16h b = *(const v16h*)(Bp + ((size_t)(kt * 8 + nt) * 32 + lane) * 16);
      acc = wmma_f16(a2[kt], b, acc);
    }
    float wv = wp2[nt * 16 + m];
#pragma unroll
    for (int r = 0; r < 8; ++r) partial[r] += fmaxf(acc[r], 0.0f) * wv;
  }

  // Reduce the 16 N-lanes of each half-wave (M is fixed per (r, half))
#pragma unroll
  for (int r = 0; r < 8; ++r) {
#pragma unroll
    for (int off = 1; off < 16; off <<= 1)
      partial[r] += __shfl_xor(partial[r], off, 32);
  }

  if (m == 0) {
    float b2 = bp2[0];
#pragma unroll
    for (int r = 0; r < 8; ++r) {
      int e = base + r + 8 * hh;
      float z = partial[r] + b2;
      float p = 1.0f / (1.0f + __expf(-z));
      outp[e] = state[e] ? p : -1000.0f;
    }
  }
}

// ------------------------------- launcher ----------------------------------

extern "C" void kernel_launch(void* const* d_in, const int* in_sizes, int n_in,
                              void* d_out, int out_size, void* d_ws, size_t ws_size,
                              hipStream_t stream) {
  (void)in_sizes; (void)n_in; (void)out_size; (void)ws_size;
  const float* x     = (const float*)d_in[0];
  const float* eattr = (const float*)d_in[1];
  const float* grep  = (const float*)d_in[2];
  const float* srep  = (const float*)d_in[3];
  const float* W0    = (const float*)d_in[4];
  const float* b0    = (const float*)d_in[5];
  const float* W1    = (const float*)d_in[6];
  const float* b1    = (const float*)d_in[7];
  const float* W2    = (const float*)d_in[8];
  const float* b2    = (const float*)d_in[9];
  const float* We    = (const float*)d_in[10];
  const float* be    = (const float*)d_in[11];
  const float* Wr    = (const float*)d_in[12];
  const float* br    = (const float*)d_in[13];
  const float* Wp1   = (const float*)d_in[14];
  const float* bp1   = (const float*)d_in[15];
  const float* wp2   = (const float*)d_in[16];
  const float* bp2   = (const float*)d_in[17];
  const int*   ei    = (const int*)d_in[18];
  const unsigned char* state = (const unsigned char*)d_in[19];
  const int* src = ei;
  const int* dst = ei + NEDGE;
  float* outp = (float*)d_out;

  // workspace carve-up (256B aligned)
  char* ws = (char*)d_ws;
  size_t off = 0;
  auto carve = [&](size_t bytes) {
    char* p = ws + off;
    off = (off + bytes + 255) & ~(size_t)255;
    return p;
  };
  float*     deg    = (float*)carve(NND * 4);
  float*     dinv   = (float*)carve(NND * 4);
  float*     hbuf   = (float*)carve((size_t)NND * HH * 4);
  float*     agg    = (float*)carve((size_t)NND * HH * 4);
  _Float16*  f16a   = (_Float16*)carve((size_t)NND * HH * 2);
  _Float16*  f16b   = (_Float16*)carve((size_t)NND * HH * 2);
  _Float16*  pw_w0  = (_Float16*)carve(HH * HH * 2);
  _Float16*  pw_w1  = (_Float16*)carve(HH * HH * 2);
  _Float16*  pw_w2  = (_Float16*)carve(HH * HH * 2);
  _Float16*  pw_r0  = (_Float16*)carve(HH * HH * 2);
  _Float16*  pw_r1  = (_Float16*)carve(HH * HH * 2);
  _Float16*  pw_p   = (_Float16*)carve(HH * HH * 2);
  _Float16*  pw_e   = (_Float16*)carve(32 * HH * 2);
  float*     wef32  = (float*)carve(16 * HH * 4);
  float*     biasp  = (float*)carve(HH * 4);
  float*     cvec   = (float*)carve(HH * 4);

  const int NH = NND * HH;  // 3,200,000 (multiple of 256)

  // degree / normalization
  k_zero<<<(NND + 255) / 256, 256, 0, stream>>>(deg, NND);
  k_deg<<<(NEDGE + 255) / 256, 256, 0, stream>>>(dst, deg);
  k_dinv<<<(NND + 255) / 256, 256, 0, stream>>>(deg, dinv);

  // folded small linear algebra + weight packing
  k_setup<<<9, 256, 0, stream>>>(We, Wr, br, be, grep, srep, Wp1, bp1,
                                 wef32, biasp, cvec);
  k_pack<<<32, 32, 0, stream>>>(W0, pw_w0, HH);
  k_pack<<<32, 32, 0, stream>>>(W1, pw_w1, HH);
  k_pack<<<32, 32, 0, stream>>>(W2, pw_w2, HH);
  k_pack<<<32, 32, 0, stream>>>(Wr,              pw_r0, HH);  // Wr[:H]
  k_pack<<<32, 32, 0, stream>>>(Wr + HH * HH,    pw_r1, HH);  // Wr[H:2H]
  k_pack<<<32, 32, 0, stream>>>(Wp1 + 2 * HH * HH, pw_p, HH); // Wp1[2H:]
  k_pack<<<8, 32, 0, stream>>>(wef32, pw_e, 16);              // We' (pad K->32)

  // x -> f16 (vectorized x4)
  k_f32tof16<<<(NH / 4) / 256, 256, 0, stream>>>(x, f16a, NH / 4);

  // 3 GCN layers
  const _Float16* pwW[3] = {pw_w0, pw_w1, pw_w2};
  const float* bL[3] = {b0, b1, b2};
  _Float16* cur = f16a;
  _Float16* nxt = f16b;
  const int NT = (NND + 15) / 16;           // 1563 tiles
  const int GB = (NT + 7) / 8;              // 196 blocks
  for (int l = 0; l < 3; ++l) {
    k_gemm_node<<<GB, 256, 0, stream>>>(cur, pwW[l], hbuf);
    k_zero<<<NH / 256, 256, 0, stream>>>(agg, NH);
    k_scatter<<<(NEDGE * (HH / 4)) / 256, 256, 0, stream>>>(hbuf, src, dst, dinv,
                                                            agg);
    k_finish<<<(NH / 4 + 255) / 256, 256, 0, stream>>>(agg, hbuf, dinv, bL[l], nxt);
    _Float16* t = cur; cur = nxt; nxt = t;
  }

  // fused edge MLP: 25000 tiles of 16 edges, 8 waves/block -> 3125 blocks
  k_edge<<<NEDGE / 128, 256, 0, stream>>>(cur, eattr, pw_r0, pw_r1, pw_e, pw_p,
                                          biasp, cvec, wp2, bp2, src, dst, state,
                                          outp);
}
